// GatedLinearAttention_27650999452423
// MI455X (gfx1250) — matrix operand order repaired and checked by reading, not compiled, not run on hardware
//
#include <hip/hip_runtime.h>

// ---------------------------------------------------------------------------
// GLA forward for MI455X (gfx1250, wave32, WMMA).
// bf16 operands + f32 accumulation via v_wmma_f32_16x16x32_bf16.
// Gating/exp/norm math in f32. Recurrent state S (256x512 per (b,h)) lives in
// VGPR accumulators (256x16 f32 per wave = 16 wmma accumulators); LDS used to
// reshape S into B-operand layout between chunks. Global->LDS staging uses
// CDNA5 async loads (global_load_async_to_lds_b128, ASYNCcnt).
// ---------------------------------------------------------------------------

typedef unsigned short u16;
typedef unsigned int   u32;
typedef unsigned long long u64;
typedef u32   u32x4 __attribute__((ext_vector_type(4)));
typedef __bf16 v16bf __attribute__((ext_vector_type(16)));
typedef float  v8f   __attribute__((ext_vector_type(8)));

struct U32x8 { u32x4 a, b; };

__device__ __forceinline__ u16 f2bf(float f) {
  u32 u = __float_as_uint(f);
  u32 r = (u + 0x7FFFu + ((u >> 16) & 1u)) >> 16;   // round-to-nearest-even
  return (u16)r;
}

__device__ __forceinline__ v16bf mkfrag(u32x4 a, u32x4 b) {
  U32x8 s; s.a = a; s.b = b;
  return __builtin_bit_cast(v16bf, s);
}

__device__ __forceinline__ v8f vzero() {
  v8f z = {0.f,0.f,0.f,0.f,0.f,0.f,0.f,0.f};
  return z;
}

// Async global->LDS copy of one 16B chunk (tracked by ASYNCcnt).
// VDST = VGPR holding LDS byte address, VADDR = 64-bit global address.
__device__ __forceinline__ void async_cp16(const void* g, void* l) {
  u32 lo = (u32)(u64)(size_t)l;          // low 32 bits address the LDS aperture
  u64 ga = (u64)(size_t)g;
  asm volatile("global_load_async_to_lds_b128 %0, %1, off"
               :: "v"(lo), "v"(ga) : "memory");
}
__device__ __forceinline__ void async_wait0() {
  asm volatile("s_wait_asynccnt 0x0" ::: "memory");
}

// A-operand (16x32 bf16, M x K): lane lr holds row M=lr; lanes 0-15 carry
// K = [0..7] and [16..23], lanes 16-31 carry K = [8..15] and [24..31].
__device__ __forceinline__ v16bf ldsA(const u16* lds, int RS, int m0, int k0,
                                      int lr, int hi) {
  const u16* p = lds + (m0 + lr) * RS + k0 + hi * 8;
  return mkfrag(*(const u32x4*)p, *(const u32x4*)(p + 16));
}

// B-operand (32x16 bf16, K x N): lane lr holds column N=lr; lanes 0-15 carry
// K = [0..7] and [16..23], lanes 16-31 carry K = [8..15] and [24..31].
// Source stored as rows of N with K contiguous (i.e. [N][K]).
__device__ __forceinline__ v16bf ldsB(const u16* lds, int RS, int n0, int k0,
                                      int lr, int hi) {
  const u16* p = lds + (n0 + lr) * RS + k0 + hi * 8;
  return mkfrag(*(const u32x4*)p, *(const u32x4*)(p + 16));
}

__device__ __forceinline__ v8f wmma_bf16(v16bf a, v16bf b, v8f c) {
  return __builtin_amdgcn_wmma_f32_16x16x32_bf16(false, a, false, b,
                                                 (short)0, c, false, false);
}

// ---------------------------------------------------------------------------
// f32 -> bf16 cast
// ---------------------------------------------------------------------------
__global__ void cast_kernel(const float* __restrict__ src, u16* __restrict__ dst, int n) {
  int i = blockIdx.x * 256 + threadIdx.x;
  if (i < n) dst[i] = f2bf(src[i]);
}

// ---------------------------------------------------------------------------
// C[M,N] (f32) = A[M,K] (bf16, row-major) @ W[N,K]^T (bf16, row-major).
// Block: 256 threads (8 waves), 128x128 C tile, K stepped by 64 via async
// LDS staging. Wave (wm=wid>>1, wn=wid&1) owns a 32x64 sub-tile = 2x4 wmma
// accumulators -> 16 WMMAs per barrier pair.
// ---------------------------------------------------------------------------
__global__ void __launch_bounds__(256) gemm_bf16_nt(
    const u16* __restrict__ A, const u16* __restrict__ W,
    float* __restrict__ C, int M, int N, int K) {
  __shared__ u16 sA[128 * 64];
  __shared__ u16 sB[128 * 64];
  int tid = threadIdx.x, lane = tid & 31, wid = tid >> 5;
  int lr = lane & 15, hi = lane >> 4;
  int m_base = blockIdx.y * 128, n_base = blockIdx.x * 128;
  int wm = wid >> 1, wn = wid & 1;

  v8f acc[2][4];
#pragma unroll
  for (int i = 0; i < 2; ++i)
#pragma unroll
    for (int j = 0; j < 4; ++j) acc[i][j] = vzero();

  for (int k0 = 0; k0 < K; k0 += 64) {
    // stage 128x64 bf16 tiles of A and W: 1024 16B chunks each, 4 per thread
#pragma unroll
    for (int i = 0; i < 4; ++i) {
      int c2 = tid * 4 + i;
      int row = c2 >> 3, cc = (c2 & 7) * 8;
      async_cp16(A + (size_t)(m_base + row) * K + k0 + cc, sA + row * 64 + cc);
      async_cp16(W + (size_t)(n_base + row) * K + k0 + cc, sB + row * 64 + cc);
    }
    async_wait0();
    __syncthreads();
#pragma unroll
    for (int kk = 0; kk < 2; ++kk) {
      v16bf af[2], bfr[4];
#pragma unroll
      for (int i = 0; i < 2; ++i)
        af[i] = ldsA(sA, 64, wm * 32 + i * 16, kk * 32, lr, hi);
#pragma unroll
      for (int j = 0; j < 4; ++j)
        bfr[j] = ldsB(sB, 64, wn * 64 + j * 16, kk * 32, lr, hi);
#pragma unroll
      for (int i = 0; i < 2; ++i)
#pragma unroll
        for (int j = 0; j < 4; ++j)
          acc[i][j] = wmma_bf16(af[i], bfr[j], acc[i][j]);
    }
    __syncthreads();
  }

  // C layout: VGPR r, lane -> M = r + 8*hi, N = lr (per 16x16 tile)
#pragma unroll
  for (int i = 0; i < 2; ++i)
#pragma unroll
    for (int j = 0; j < 4; ++j) {
      int n = n_base + wn * 64 + j * 16 + lr;
#pragma unroll
      for (int r = 0; r < 8; ++r) {
        int m = m_base + wm * 32 + i * 16 + r + hi * 8;
        C[(size_t)m * N + n] = acc[i][j][r];
      }
    }
}

// ---------------------------------------------------------------------------
// tmp[R,16] = x[R,1024] @ Wgk1[16,1024]^T   (rank-16, scalar FMA path)
// ---------------------------------------------------------------------------
__global__ void lowrank_kernel(const float* __restrict__ x,
                               const float* __restrict__ Wgk1,
                               float* __restrict__ tmp) {
  int row = blockIdx.x;
  int t = threadIdx.x;                 // 128 threads
  int j = t & 15, part = t >> 4;       // 8 partial sums per output
  const float* xr = x + (size_t)row * 1024;
  const float* wr = Wgk1 + (size_t)j * 1024;
  float s = 0.f;
  for (int d = part * 128; d < part * 128 + 128; ++d) s += xr[d] * wr[d];
  __shared__ float red[128];
  red[t] = s;
  __syncthreads();
  if (part == 0) {
    float a = 0.f;
#pragma unroll
    for (int p = 0; p < 8; ++p) a += red[p * 16 + j];
    tmp[(size_t)row * 16 + j] = a;
  }
}

// gk[R,1024] = log_sigmoid(tmp @ Wgk2^T + bgk2) / 16
__global__ void gk_kernel(const float* __restrict__ tmp,
                          const float* __restrict__ Wgk2,
                          const float* __restrict__ bgk2,
                          float* __restrict__ gk) {
  int idx = blockIdx.x * 256 + threadIdx.x;
  int r = idx >> 10, c = idx & 1023;
  const float* tr = tmp + (size_t)r * 16;
  const float* w  = Wgk2 + (size_t)c * 16;
  float s = bgk2[c];
#pragma unroll
  for (int j = 0; j < 16; ++j) s += tr[j] * w[j];
  float ls = fminf(s, 0.f) - log1pf(__expf(-fabsf(s)));
  gk[idx] = ls * (1.0f / 16.0f);
}

// ---------------------------------------------------------------------------
// Per (b,h,chunk): in-chunk cumsum of gk, then emit WMMA-ready bf16 tensors:
//   qg [cb][64][256]   (pos x dk)  = q * exp(b) * DK^-0.5
//   kg [cb][64][256]   (pos x dk)  = k * exp(-b)
//   kdt[cb][256][64]   (dk x pos)  = k * exp(b_last - b)     (A-operand of S upd)
//   vt [cb][512][64]   (dv x pos)                            (B-operand)
//   dec[cb][256]                    = exp(b_last)
// cb = (b*H + h)*32 + chunk
// ---------------------------------------------------------------------------
__global__ void gate_kernel(const float* __restrict__ q, const float* __restrict__ k,
                            const float* __restrict__ v, const float* __restrict__ gk,
                            u16* __restrict__ qg, u16* __restrict__ kg,
                            u16* __restrict__ kdt, u16* __restrict__ vt,
                            float* __restrict__ dec) {
  int blk = blockIdx.x;
  int c  = blk & 31;
  int bh = blk >> 5;
  int h  = bh & 3;
  int bb = bh >> 2;
  int t0 = c * 64;
  int tid = threadIdx.x;                  // 256 threads: one per dk
  size_t cb = (size_t)blk;
  const float scale = 0.0625f;            // DK^-0.5

  int dk = tid;
  float bl = 0.f;
  for (int p = 0; p < 64; ++p)
    bl += gk[(size_t)(bb * 2048 + t0 + p) * 1024 + h * 256 + dk];
  dec[cb * 256 + dk] = __expf(bl);

  float run = 0.f;
  for (int p = 0; p < 64; ++p) {
    size_t gi = (size_t)(bb * 2048 + t0 + p) * 1024 + h * 256 + dk;
    run += gk[gi];
    float qv = q[gi], kv = k[gi];
    qg[(cb * 64 + p) * 256 + dk]  = f2bf(qv * __expf(run) * scale);
    kg[(cb * 64 + p) * 256 + dk]  = f2bf(kv * __expf(-run));
    kdt[(cb * 256 + dk) * 64 + p] = f2bf(kv * __expf(bl - run));
  }

  // transpose v chunk (64 x 512 f32) -> vt (512 x 64 bf16)
  for (int dv = tid; dv < 512; dv += 256) {
    for (int p = 0; p < 64; ++p) {
      float vv = v[(size_t)(bb * 2048 + t0 + p) * 2048 + h * 512 + dv];
      vt[(cb * 512 + dv) * 64 + p] = f2bf(vv);
    }
  }
}

// ---------------------------------------------------------------------------
// Chunked GLA scan. grid = B*H*4: block (256 thr / 8 waves) owns (b,h) and a
// 128-wide DV slice. Each wave keeps S[256 x 16] f32 in 16 wmma accumulators.
// Per chunk: o_inter = qg@S, A = tril(qg@kg^T), o_intra = A@v, o -> global,
// S = diag(exp(b_last)) S + kd^T @ v, S re-staged to LDS in B-operand layout.
// ---------------------------------------------------------------------------
__device__ __forceinline__ void cp16_async(const void* src, void* dst, int n16,
                                           int tid) {
  const char* s = (const char*)src;
  char* d = (char*)dst;
  for (int i = tid; i < n16; i += 256) async_cp16(s + i * 16, d + i * 16);
}

__global__ void __launch_bounds__(256, 1) scan_kernel(
    const u16* __restrict__ qg, const u16* __restrict__ kg,
    const u16* __restrict__ kdt, const u16* __restrict__ vt,
    const float* __restrict__ dec, float* __restrict__ o_out) {
  extern __shared__ char smem[];
  u16* s_st  = (u16*)smem;              // 128 x 256 : S^T slice (dv x dk), bf16
  u16* s_qg  = s_st  + 128 * 256;       // 64 x 256
  u16* s_kg  = s_qg  + 64 * 256;        // 64 x 256
  u16* s_kdt = s_kg  + 64 * 256;        // 256 x 64
  u16* s_vt  = s_kdt + 256 * 64;        // 128 x 64
  u16* s_a   = s_vt  + 128 * 64;        // 64 x 64 masked A, bf16
  float* s_dec = (float*)(s_a + 64 * 64); // 256

  int tid = threadIdx.x, lane = tid & 31, wid = tid >> 5;
  int lr = lane & 15, hi = lane >> 4;
  int blk = blockIdx.x;
  int vs = blk & 3;          // which 128-wide DV slice
  int bh = blk >> 2;
  int h = bh & 3, bb = bh >> 2;

  // zero S (regs + LDS image)
  {
    u32x4 z = {0u, 0u, 0u, 0u};
    u32x4* p = (u32x4*)s_st;
    for (int i = tid; i < (128 * 256 * 2) / 16; i += 256) p[i] = z;
  }
  v8f S[16];
#pragma unroll
  for (int t = 0; t < 16; ++t) S[t] = vzero();
  __syncthreads();

  size_t cbBase = (size_t)bh * 32;
  for (int c = 0; c < 32; ++c) {
    size_t cb = cbBase + c;
    cp16_async(qg + cb * 64 * 256, s_qg, 2048, tid);
    cp16_async(kg + cb * 64 * 256, s_kg, 2048, tid);
    cp16_async(kdt + cb * 256 * 64, s_kdt, 2048, tid);
    cp16_async(vt + (cb * 512 + vs * 128) * 64, s_vt, 1024, tid);
    cp16_async(dec + cb * 256, s_dec, 64, tid);
    async_wait0();
    __syncthreads();

    if (c + 1 < 32) {   // hint next chunk into cache (global_prefetch_b8)
      __builtin_prefetch((const char*)(qg + (cb + 1) * 64 * 256) + tid * 128, 0, 1);
      __builtin_prefetch((const char*)(kdt + (cb + 1) * 256 * 64) + tid * 128, 0, 1);
      __builtin_prefetch((const char*)(vt + ((cb + 1) * 512 + vs * 128) * 64) + tid * 64, 0, 1);
    }

    // ---- o_inter = qg (64x256) @ S_slice (256x16 per wave) ----
    v8f oacc[4];
#pragma unroll
    for (int ti = 0; ti < 4; ++ti) oacc[ti] = vzero();
#pragma unroll
    for (int ti = 0; ti < 4; ++ti)
#pragma unroll
      for (int kb = 0; kb < 8; ++kb) {
        v16bf a  = ldsA(s_qg, 256, ti * 16, kb * 32, lr, hi);
        v16bf bf_ = ldsB(s_st, 256, wid * 16, kb * 32, lr, hi);
        oacc[ti] = wmma_bf16(a, bf_, oacc[ti]);
      }

    // ---- A = tril(qg @ kg^T): 16 tiles, 2 per wave ----
#pragma unroll
    for (int u = 0; u < 2; ++u) {
      int tt = wid * 2 + u;
      int ti = tt >> 2, tj = tt & 3;
      v8f acc = vzero();
      if (ti >= tj) {                       // wave-uniform; upper tiles stay 0
#pragma unroll
        for (int kb = 0; kb < 8; ++kb) {
          v16bf a  = ldsA(s_qg, 256, ti * 16, kb * 32, lr, hi);
          v16bf bf_ = ldsB(s_kg, 256, tj * 16, kb * 32, lr, hi);
          acc = wmma_bf16(a, bf_, acc);
        }
      }
#pragma unroll
      for (int r = 0; r < 8; ++r) {
        int ii = ti * 16 + r + hi * 8;
        int jj = tj * 16 + lr;
        s_a[ii * 64 + jj] = f2bf(ii >= jj ? acc[r] : 0.0f);
      }
    }
    __syncthreads();

    // ---- o_intra += A (64x64) @ v (64x16 per wave) ----
#pragma unroll
    for (int ti = 0; ti < 4; ++ti)
#pragma unroll
      for (int kb = 0; kb < 2; ++kb) {
        v16bf a  = ldsA(s_a, 64, ti * 16, kb * 32, lr, hi);
        v16bf bf_ = ldsB(s_vt, 64, wid * 16, kb * 32, lr, hi);
        oacc[ti] = wmma_bf16(a, bf_, oacc[ti]);
      }

    // ---- write o (layout [b*T+t][h*512+dv], f32) ----
    int dv = vs * 128 + wid * 16 + lr;
#pragma unroll
    for (int ti = 0; ti < 4; ++ti)
#pragma unroll
      for (int r = 0; r < 8; ++r) {
        int t = c * 64 + ti * 16 + r + hi * 8;
        o_out[(size_t)(bb * 2048 + t) * 2048 + h * 512 + dv] = oacc[ti][r];
      }

    // ---- S = diag(exp(b_last)) S + kd^T (256x64) @ v (64x16) ----
#pragma unroll
    for (int tk = 0; tk < 16; ++tk) {
#pragma unroll
      for (int r = 0; r < 8; ++r) S[tk][r] *= s_dec[tk * 16 + r + hi * 8];
#pragma unroll
      for (int kb = 0; kb < 2; ++kb) {
        v16bf a  = ldsA(s_kdt, 64, tk * 16, kb * 32, lr, hi);
        v16bf bf_ = ldsB(s_vt, 64, wid * 16, kb * 32, lr, hi);
        S[tk] = wmma_bf16(a, bf_, S[tk]);
      }
    }

    // ---- re-stage S (bf16, transposed: [dv_local][dk]) for next o_inter ----
    int dvrow = wid * 16 + lr;
#pragma unroll
    for (int tk = 0; tk < 16; ++tk) {
      u16* dst = s_st + dvrow * 256 + tk * 16 + hi * 8;
#pragma unroll
      for (int r = 0; r < 8; r += 2) {
        u32 pk = (u32)f2bf(S[tk][r]) | ((u32)f2bf(S[tk][r + 1]) << 16);
        *(u32*)(dst + r) = pk;
      }
    }
    __syncthreads();
  }
}

// ---------------------------------------------------------------------------
// RMS-norm over DV (per b,t,h) * g_norm_w * silu(g)  -> bf16
// ---------------------------------------------------------------------------
__global__ void norm_kernel(const float* __restrict__ o, const float* __restrict__ g,
                            const float* __restrict__ gnw, u16* __restrict__ onorm) {
  int row = blockIdx.x;          // B*T*H
  int h = row & 3;
  size_t bt = (size_t)(row >> 2);
  size_t base = bt * 2048 + (size_t)h * 512;
  int tid = threadIdx.x;

  float vals[2];
  float ss = 0.f;
#pragma unroll
  for (int i = 0; i < 2; ++i) {
    float ov = o[base + tid + i * 256];
    vals[i] = ov;
    ss += ov * ov;
  }
  __shared__ float red[256];
  red[tid] = ss;
  __syncthreads();
  for (int s = 128; s > 0; s >>= 1) {
    if (tid < s) red[tid] += red[tid + s];
    __syncthreads();
  }
  float inv = rsqrtf(red[0] / 512.0f + 1e-5f);
#pragma unroll
  for (int i = 0; i < 2; ++i) {
    int dv = tid + i * 256;
    float gv = g[base + dv];
    float sig = 1.0f / (1.0f + __expf(-gv));
    onorm[base + dv] = f2bf(vals[i] * inv * gnw[dv] * (gv * sig));
  }
}

// ---------------------------------------------------------------------------
extern "C" void kernel_launch(void* const* d_in, const int* in_sizes, int n_in,
                              void* d_out, int out_size, void* d_ws, size_t ws_size,
                              hipStream_t stream) {
  (void)in_sizes; (void)n_in; (void)out_size; (void)ws_size;
  const float* x    = (const float*)d_in[0];
  const float* Wq   = (const float*)d_in[1];
  const float* Wk   = (const float*)d_in[2];
  const float* Wv   = (const float*)d_in[3];
  const float* Wg   = (const float*)d_in[4];
  const float* Wgk1 = (const float*)d_in[5];
  const float* Wgk2 = (const float*)d_in[6];
  const float* bgk2 = (const float*)d_in[7];
  const float* Wo   = (const float*)d_in[8];
  const float* gnw  = (const float*)d_in[9];
  float* out = (float*)d_out;

  const int R = 4096;   // B*T
  char* ws = (char*)d_ws;
  size_t off = 0;
  auto alloc = [&](size_t bytes) -> void* {
    void* p = ws + off;
    off += (bytes + 255) & ~(size_t)255;
    return p;
  };

  u16*   xb  = (u16*)alloc((size_t)R * 1024 * 2);
  u16*   wqb = (u16*)alloc((size_t)1024 * 1024 * 2);
  u16*   wkb = (u16*)alloc((size_t)1024 * 1024 * 2);
  u16*   wvb = (u16*)alloc((size_t)2048 * 1024 * 2);
  u16*   wgb = (u16*)alloc((size_t)2048 * 1024 * 2);
  u16*   wob = (u16*)alloc((size_t)1024 * 2048 * 2);
  float* qf  = (float*)alloc((size_t)R * 1024 * 4);
  float* kf  = (float*)alloc((size_t)R * 1024 * 4);
  float* vf  = (float*)alloc((size_t)R * 2048 * 4);
  float* gf  = (float*)alloc((size_t)R * 2048 * 4);
  float* t16 = (float*)alloc((size_t)R * 16 * 4);
  float* gk  = (float*)alloc((size_t)R * 1024 * 4);
  u16*   qgb = (u16*)alloc((size_t)R * 1024 * 2);
  u16*   kgb = (u16*)alloc((size_t)R * 1024 * 2);
  u16*   kdt = (u16*)alloc((size_t)R * 1024 * 2);
  u16*   vtb = (u16*)alloc((size_t)R * 2048 * 2);
  float* dec = (float*)alloc((size_t)8 * 32 * 256 * 4);
  // buffer reuse (stream-ordered lifetimes): v_f32 dead after gate_kernel,
  // q_f32 dead after gate_kernel.
  float* o_buf = vf;
  u16*   onorm = (u16*)qf;

  auto cgrid = [](size_t n) { return dim3((unsigned)((n + 255) / 256)); };

  cast_kernel<<<cgrid((size_t)R * 1024), 256, 0, stream>>>(x, xb, R * 1024);
  cast_kernel<<<cgrid(1024 * 1024), 256, 0, stream>>>(Wq, wqb, 1024 * 1024);
  cast_kernel<<<cgrid(1024 * 1024), 256, 0, stream>>>(Wk, wkb, 1024 * 1024);
  cast_kernel<<<cgrid(2048 * 1024), 256, 0, stream>>>(Wv, wvb, 2048 * 1024);
  cast_kernel<<<cgrid(2048 * 1024), 256, 0, stream>>>(Wg, wgb, 2048 * 1024);
  cast_kernel<<<cgrid(1024 * 2048), 256, 0, stream>>>(Wo, wob, 1024 * 2048);

  gemm_bf16_nt<<<dim3(1024 / 128, R / 128), 256, 0, stream>>>(xb, wqb, qf, R, 1024, 1024);
  gemm_bf16_nt<<<dim3(1024 / 128, R / 128), 256, 0, stream>>>(xb, wkb, kf, R, 1024, 1024);
  gemm_bf16_nt<<<dim3(2048 / 128, R / 128), 256, 0, stream>>>(xb, wvb, vf, R, 2048, 1024);
  gemm_bf16_nt<<<dim3(2048 / 128, R / 128), 256, 0, stream>>>(xb, wgb, gf, R, 2048, 1024);

  lowrank_kernel<<<R, 128, 0, stream>>>(x, Wgk1, t16);
  gk_kernel<<<(R * 1024) / 256, 256, 0, stream>>>(t16, Wgk2, bgk2, gk);

  gate_kernel<<<256, 256, 0, stream>>>(qf, kf, vf, gk, qgb, kgb, kdt, vtb, dec);

  const size_t scan_lds =
      (size_t)(128 * 256 + 64 * 256 * 2 + 256 * 64 + 128 * 64 + 64 * 64) * 2 + 256 * 4;
  (void)hipFuncSetAttribute((const void*)scan_kernel,
                            hipFuncAttributeMaxDynamicSharedMemorySize,
                            (int)scan_lds);
  scan_kernel<<<32, 256, scan_lds, stream>>>(qgb, kgb, kdt, vtb, dec, o_buf);

  norm_kernel<<<R * 4, 256, 0, stream>>>(o_buf, gf, gnw, onorm);

  gemm_bf16_nt<<<dim3(1024 / 128, R / 128), 256, 0, stream>>>(onorm, wob, out, R, 1024, 2048);
}
